// AdaptiveGCN_77060303225421
// MI455X (gfx1250) — matrix operand order repaired and checked by reading, not compiled, not run on hardware
//
#include <hip/hip_runtime.h>
#include <cstdint>
#include <cstddef>

// ---------------------------------------------------------------------------
// Types for CDNA5 WMMA (wave32): bf16 A/B fragments (v16), f32 accum (v8).
// ---------------------------------------------------------------------------
typedef __bf16 bf16;
typedef __attribute__((ext_vector_type(16))) __bf16 v16bf;
typedef __attribute__((ext_vector_type(8)))  __bf16 v8bf;
typedef __attribute__((ext_vector_type(8)))  float  v8f;

// Pointer types for the async global->LDS builtin (per hipcc diagnostic:
// param is 'int __attribute__((vector_size(16)))*', generic AS for source).
typedef int v4i __attribute__((vector_size(16)));
#if defined(__HIP_DEVICE_COMPILE__)
typedef __attribute__((address_space(3))) v4i lds_v4i;
#endif

static constexpr int Nn   = 4096;  // nodes
static constexpr int CIN  = 1024;
static constexpr int HH   = 1024;
static constexpr int COUT = 512;
static constexpr int RNK  = 10;

// CDNA5 async global->LDS staging (ASYNCcnt) if the toolchain exposes it.
#if defined(__HIP_DEVICE_COMPILE__) && \
    __has_builtin(__builtin_amdgcn_global_load_async_to_lds_b128) && \
    __has_builtin(__builtin_amdgcn_s_wait_asynccnt)
#define USE_ASYNC_LDS 1
#else
#define USE_ASYNC_LDS 0
#endif

// ---------------------------------------------------------------------------
// fp32 -> bf16 convert (row-major preserved)
// ---------------------------------------------------------------------------
__global__ void k_cvt_bf16(const float* __restrict__ s, bf16* __restrict__ d, int n) {
  int i = blockIdx.x * blockDim.x + threadIdx.x;
  if (i < n) d[i] = (bf16)s[i];
}

// ---------------------------------------------------------------------------
// fp32 [R][Cc] -> bf16 [Cc][R] tiled transpose-convert (32x32 tiles via LDS)
// ---------------------------------------------------------------------------
__global__ __launch_bounds__(256)
void k_cvt_t_bf16(const float* __restrict__ s, bf16* __restrict__ d, int R, int Cc) {
  __shared__ float tile[32][33];
  const int bx = blockIdx.x;            // over Cc/32
  const int by = blockIdx.y;            // over R/32
  const int tx = threadIdx.x & 31, ty = threadIdx.x >> 5;  // 32 x 8
#pragma unroll
  for (int r = ty; r < 32; r += 8)
    tile[r][tx] = s[(size_t)(by * 32 + r) * Cc + bx * 32 + tx];
  __syncthreads();
#pragma unroll
  for (int r = ty; r < 32; r += 8)
    d[(size_t)(bx * 32 + r) * R + by * 32 + tx] = (bf16)tile[tx][r];
}

// ---------------------------------------------------------------------------
// Per-row softmax stats of relu(nv1 @ nv2): rowmax[i], rowinv[i] = 1/rowsum.
// ---------------------------------------------------------------------------
__global__ __launch_bounds__(256)
void k_adj_row(const float* __restrict__ nv1, const float* __restrict__ nv2,
               float* __restrict__ rowmax, float* __restrict__ rowinv) {
  __shared__ float red[256];
  const int i = blockIdx.x, t = threadIdx.x;
  float a[RNK];
#pragma unroll
  for (int r = 0; r < RNK; ++r) a[r] = nv1[i * RNK + r];

  float mx = 0.0f;  // relu output >= 0
  for (int j = t; j < Nn; j += 256) {
    float l = 0.0f;
#pragma unroll
    for (int r = 0; r < RNK; ++r) l += a[r] * nv2[r * Nn + j];
    mx = fmaxf(mx, fmaxf(l, 0.0f));
  }
  red[t] = mx; __syncthreads();
  for (int s = 128; s > 0; s >>= 1) { if (t < s) red[t] = fmaxf(red[t], red[t + s]); __syncthreads(); }
  mx = red[0]; __syncthreads();

  float sm = 0.0f;
  for (int j = t; j < Nn; j += 256) {
    float l = 0.0f;
#pragma unroll
    for (int r = 0; r < RNK; ++r) l += a[r] * nv2[r * Nn + j];
    sm += __expf(fmaxf(l, 0.0f) - mx);
  }
  red[t] = sm; __syncthreads();
  for (int s = 128; s > 0; s >>= 1) { if (t < s) red[t] += red[t + s]; __syncthreads(); }
  if (t == 0) { rowmax[i] = mx; rowinv[i] = 1.0f / red[0]; }
}

// ---------------------------------------------------------------------------
// Column degree of A = adp + I, then dinv[j] = rsqrt(deg). Atomic-free.
// ---------------------------------------------------------------------------
__global__ __launch_bounds__(256)
void k_col_dinv(const float* __restrict__ nv1, const float* __restrict__ nv2,
                const float* __restrict__ rowmax, const float* __restrict__ rowinv,
                float* __restrict__ dinv) {
  int j = blockIdx.x * 256 + threadIdx.x;
  if (j >= Nn) return;
  float d = 1.0f;  // self-loop
  for (int i = 0; i < Nn; ++i) {
    float l = 0.0f;
#pragma unroll
    for (int r = 0; r < RNK; ++r) l += nv1[i * RNK + r] * nv2[r * Nn + j];
    d += __expf(fmaxf(l, 0.0f) - rowmax[i]) * rowinv[i];
  }
  dinv[j] = rsqrtf(d);
}

// ---------------------------------------------------------------------------
// Build Bn = An^T in bf16 row-major: Bn[i][j] = dinv[i]*dinv[j]*(adp[j][i]+dij)
// ---------------------------------------------------------------------------
__global__ __launch_bounds__(256)
void k_build_adj(const float* __restrict__ nv1, const float* __restrict__ nv2,
                 const float* __restrict__ rowmax, const float* __restrict__ rowinv,
                 const float* __restrict__ dinv, bf16* __restrict__ Bn) {
  const int t = threadIdx.x;
  const int j = blockIdx.x * 32 + (t & 31);
  const int i = blockIdx.y * 8 + (t >> 5);
  float l = 0.0f;
#pragma unroll
  for (int r = 0; r < RNK; ++r) l += nv1[j * RNK + r] * nv2[r * Nn + i];
  float adp = __expf(fmaxf(l, 0.0f) - rowmax[j]) * rowinv[j];
  float v = dinv[i] * dinv[j] * (adp + (i == j ? 1.0f : 0.0f));
  Bn[(size_t)i * Nn + j] = (bf16)v;
}

// ---------------------------------------------------------------------------
// Epilogue vector store helpers (8 consecutive elements, one lane)
// ---------------------------------------------------------------------------
__device__ inline void store8(bf16* p, const float* v) {
  v8bf o;
#pragma unroll
  for (int r = 0; r < 8; ++r) o[r] = (bf16)v[r];
  *(v8bf*)p = o;
}
__device__ inline void store8(float* p, const float* v) {
#pragma unroll
  for (int r = 0; r < 8; ++r) p[r] = v[r];
}

// ---------------------------------------------------------------------------
// WMMA GEMM: C = epilogue(A[M,K] @ B[K,N]),  A row-major, B given TRANSPOSED
// as BT[N][K] so both operands stage & fragment-load as contiguous b128.
// Block tile 128x128, K-tile 64. 8 waves in 4(M)x2(N); wave tile 32x64 =
// 2x4 fragments -> 8 v_wmma_f32_16x16x32_bf16 per 32-K substep.
// Staging: async global->LDS (ASYNCcnt) when available, else pipelined
// b128 loads through distinct temps (keeps 8 loads in flight).
// ---------------------------------------------------------------------------
template <typename OutT>
__global__ __launch_bounds__(256)
void k_gemm_bf16(const bf16* __restrict__ A, const bf16* __restrict__ BT,
                 OutT* __restrict__ C, int M, int N, int K,
                 const float* __restrict__ bias, int doRelu, int storeT,
                 const float* __restrict__ g, const float* __restrict__ bb,
                 const float* __restrict__ bm, const float* __restrict__ bv) {
  __shared__ __align__(16) bf16 As[128][72];  // [m][k], padded
  __shared__ __align__(16) bf16 Bs[128][72];  // [n][k], padded

  const int tid  = threadIdx.x;
  const int m0   = blockIdx.y * 128;
  const int n0   = blockIdx.x * 128;
  const int wid  = tid >> 5;
  const int lane = tid & 31;
  const int lm   = lane & 15;     // M (A-frag) / N (B-frag) index
  const int lh   = lane >> 4;     // K-half selector
  const int wm   = (wid & 3) * 32;
  const int wn   = (wid >> 2) * 64;

  // staging: each thread owns (row = tid/8 + it*32, col = (tid%8)*8)
  const int srow = tid >> 3;
  const int scol = (tid & 7) << 3;
  const bf16* aptr = A  + (size_t)(m0 + srow) * K + scol;
  const bf16* bptr = BT + (size_t)(n0 + srow) * K + scol;

  v8f acc[2][4] = {};

  for (int k0 = 0; k0 < K; k0 += 64) {
#if USE_ASYNC_LDS
    // CDNA5 async copy: memory -> LDS without VGPR staging (ASYNCcnt).
#pragma unroll
    for (int it = 0; it < 4; ++it)
      __builtin_amdgcn_global_load_async_to_lds_b128(
          (v4i*)(aptr + (size_t)it * 32 * K + k0),
          (lds_v4i*)&As[srow + it * 32][scol], 0, 0);
#pragma unroll
    for (int it = 0; it < 4; ++it)
      __builtin_amdgcn_global_load_async_to_lds_b128(
          (v4i*)(bptr + (size_t)it * 32 * K + k0),
          (lds_v4i*)&Bs[srow + it * 32][scol], 0, 0);
#else
    // Pipelined staging: 8 b128 loads in flight, then 8 LDS stores.
    v8bf ta[4], tb[4];
#pragma unroll
    for (int it = 0; it < 4; ++it) ta[it] = *(const v8bf*)(aptr + (size_t)it * 32 * K + k0);
#pragma unroll
    for (int it = 0; it < 4; ++it) tb[it] = *(const v8bf*)(bptr + (size_t)it * 32 * K + k0);
#pragma unroll
    for (int it = 0; it < 4; ++it) *(v8bf*)&As[srow + it * 32][scol] = ta[it];
#pragma unroll
    for (int it = 0; it < 4; ++it) *(v8bf*)&Bs[srow + it * 32][scol] = tb[it];
#endif
    if (k0 + 64 < K) {  // pull next K tile toward the caches
      __builtin_prefetch(aptr + k0 + 64, 0, 1);
      __builtin_prefetch(bptr + k0 + 64, 0, 1);
    }
#if USE_ASYNC_LDS
    __builtin_amdgcn_s_wait_asynccnt(0);
#endif
    __syncthreads();

#pragma unroll
    for (int ks = 0; ks < 64; ks += 32) {
      v16bf af[2], bfr[4];
#pragma unroll
      for (int mi = 0; mi < 2; ++mi) {
        const bf16* pA = &As[wm + mi * 16 + lm][ks + lh * 8];
        af[mi] = __builtin_shufflevector(*(const v8bf*)pA, *(const v8bf*)(pA + 16),
                                         0,1,2,3,4,5,6,7,8,9,10,11,12,13,14,15);
      }
#pragma unroll
      for (int ni = 0; ni < 4; ++ni) {
        const bf16* pB = &Bs[wn + ni * 16 + lm][ks + lh * 8];
        bfr[ni] = __builtin_shufflevector(*(const v8bf*)pB, *(const v8bf*)(pB + 16),
                                          0,1,2,3,4,5,6,7,8,9,10,11,12,13,14,15);
      }
#pragma unroll
      for (int mi = 0; mi < 2; ++mi)
#pragma unroll
        for (int ni = 0; ni < 4; ++ni)
          acc[mi][ni] = __builtin_amdgcn_wmma_f32_16x16x32_bf16(
              false, af[mi], false, bfr[ni], (short)0, acc[mi][ni], false, false);
    }
    __syncthreads();
  }

  // ---- epilogue: D VGPR r -> M = (lane/16)*8 + r, N = lane%16 ----
#pragma unroll
  for (int ni = 0; ni < 4; ++ni) {
    const int col = n0 + wn + ni * 16 + lm;
    const float bi = bias ? bias[col] : 0.0f;
    float scale = 1.0f, shift = 0.0f;
    if (g) {  // eval BatchNorm folded to scale+shift
      scale = rsqrtf(bv[col] + 1e-5f) * g[col];
      shift = bb[col] - bm[col] * scale;
    }
#pragma unroll
    for (int mi = 0; mi < 2; ++mi) {
      const int rb = m0 + wm + mi * 16 + lh * 8;
      float tv[8];
#pragma unroll
      for (int r = 0; r < 8; ++r) {
        float v = acc[mi][ni][r] + bi;
        if (doRelu) v = fmaxf(v, 0.0f);
        if (g) v = v * scale + shift;
        tv[r] = v;
      }
      if (storeT) {
        store8(&C[(size_t)col * M + rb], tv);  // contiguous: one b128 (bf16)
      } else {
#pragma unroll
        for (int r = 0; r < 8; ++r) C[(size_t)(rb + r) * N + col] = (OutT)tv[r];
      }
    }
  }
}

// ---------------------------------------------------------------------------
// Mean pool over rows: pooled[c] = mean_r H2[r][c]. Coalesced across lanes.
// ---------------------------------------------------------------------------
__global__ __launch_bounds__(256)
void k_pool(const float* __restrict__ H2, float* __restrict__ pooled) {
  int c = blockIdx.x * 256 + threadIdx.x;
  if (c >= COUT) return;
  float s = 0.0f;
  for (int r = 0; r < Nn; ++r) s += H2[(size_t)r * COUT + c];
  pooled[c] = s * (1.0f / (float)Nn);
}

// ---------------------------------------------------------------------------
// Final: attn = sigmoid(pooled . w_attn + b); out = pooled * attn.
// ---------------------------------------------------------------------------
__global__ __launch_bounds__(256)
void k_final(const float* __restrict__ pooled, const float* __restrict__ w_attn,
             const float* __restrict__ b_attn, float* __restrict__ out) {
  __shared__ float red[256];
  const int t = threadIdx.x;
  float p = 0.0f;
  for (int c = t; c < COUT; c += 256) p += pooled[c] * w_attn[c];
  red[t] = p; __syncthreads();
  for (int s = 128; s > 0; s >>= 1) { if (t < s) red[t] += red[t + s]; __syncthreads(); }
  const float attn = 1.0f / (1.0f + __expf(-(red[0] + b_attn[0])));
  for (int c = t; c < COUT; c += 256) out[c] = pooled[c] * attn;
}

// ---------------------------------------------------------------------------
// Host driver
// ---------------------------------------------------------------------------
extern "C" void kernel_launch(void* const* d_in, const int* in_sizes, int n_in,
                              void* d_out, int out_size, void* d_ws, size_t ws_size,
                              hipStream_t stream) {
  (void)in_sizes; (void)n_in; (void)out_size; (void)ws_size;
  const float* x      = (const float*)d_in[0];
  const float* w_map  = (const float*)d_in[1];
  const float* b_map  = (const float*)d_in[2];
  const float* nv1    = (const float*)d_in[3];
  const float* nv2    = (const float*)d_in[4];
  const float* w1     = (const float*)d_in[5];
  const float* b1     = (const float*)d_in[6];
  const float* w2     = (const float*)d_in[7];
  const float* b2     = (const float*)d_in[8];
  const float* bn1_g  = (const float*)d_in[9];
  const float* bn1_b  = (const float*)d_in[10];
  const float* bn1_m  = (const float*)d_in[11];
  const float* bn1_v  = (const float*)d_in[12];
  const float* bn2_g  = (const float*)d_in[13];
  const float* bn2_b  = (const float*)d_in[14];
  const float* bn2_m  = (const float*)d_in[15];
  const float* bn2_v  = (const float*)d_in[16];
  const float* w_attn = (const float*)d_in[17];
  const float* b_attn = (const float*)d_in[18];
  float* out = (float*)d_out;

  // ---- workspace carve-out (~81 MB) ----
  char* p = (char*)d_ws;
  auto take = [&](size_t bytes) -> void* {
    void* q = (void*)p; p += (bytes + 255) & ~(size_t)255; return q;
  };
  bf16* Xb    = (bf16*)take((size_t)Nn * CIN * 2);    // x, bf16 row-major
  bf16* WmbT  = (bf16*)take((size_t)CIN * HH * 2);    // w_map^T  [HH][CIN]
  bf16* W1T   = (bf16*)take((size_t)HH * HH * 2);     // w1^T     [HH][HH]
  bf16* W2T   = (bf16*)take((size_t)HH * COUT * 2);   // w2^T     [COUT][HH]
  bf16* XM    = (bf16*)take((size_t)Nn * HH * 2);     // relu(x@w_map+b)
  bf16* T1T   = (bf16*)take((size_t)Nn * HH * 2);     // (XM@w1)^T [HH][Nn]
  bf16* H1    = (bf16*)take((size_t)Nn * HH * 2);     // bn1(relu(Bn@T1+b1))
  bf16* T2T   = (bf16*)take((size_t)Nn * COUT * 2);   // (H1@w2)^T [COUT][Nn]
  float* H2   = (float*)take((size_t)Nn * COUT * 4);  // bn2(relu(Bn@T2+b2))
  bf16* Bn    = (bf16*)take((size_t)Nn * Nn * 2);     // An^T  (32 MB)
  float* rowmax = (float*)take((size_t)Nn * 4);
  float* rowinv = (float*)take((size_t)Nn * 4);
  float* dinv   = (float*)take((size_t)Nn * 4);
  float* pooled = (float*)take((size_t)COUT * 4);

  // ---- precision demotion fp32 -> bf16 (weights transposed to [N][K]) ----
  k_cvt_bf16<<<(Nn * CIN + 255) / 256, 256, 0, stream>>>(x, Xb, Nn * CIN);
  k_cvt_t_bf16<<<dim3(HH / 32, CIN / 32), 256, 0, stream>>>(w_map, WmbT, CIN, HH);
  k_cvt_t_bf16<<<dim3(HH / 32, HH / 32), 256, 0, stream>>>(w1, W1T, HH, HH);
  k_cvt_t_bf16<<<dim3(COUT / 32, HH / 32), 256, 0, stream>>>(w2, W2T, HH, COUT);

  // ---- adjacency (softmax stats, degree, normalized transpose) ----
  k_adj_row<<<Nn, 256, 0, stream>>>(nv1, nv2, rowmax, rowinv);
  k_col_dinv<<<Nn / 256, 256, 0, stream>>>(nv1, nv2, rowmax, rowinv, dinv);
  k_build_adj<<<dim3(Nn / 32, Nn / 8), 256, 0, stream>>>(nv1, nv2, rowmax, rowinv, dinv, Bn);

  // ---- GEMM chain (WMMA bf16, f32 accumulate) ----
  // XM = relu(Xb @ w_map + b_map)                     [4096,1024]
  k_gemm_bf16<bf16><<<dim3(HH / 128, Nn / 128), 256, 0, stream>>>(
      Xb, WmbT, XM, Nn, HH, CIN, b_map, 1, 0, nullptr, nullptr, nullptr, nullptr);
  // T1T = (XM @ w1)^T                                 [1024,4096]
  k_gemm_bf16<bf16><<<dim3(HH / 128, Nn / 128), 256, 0, stream>>>(
      XM, W1T, T1T, Nn, HH, HH, nullptr, 0, 1, nullptr, nullptr, nullptr, nullptr);
  // H1 = bn1(relu(Bn @ T1 + b1))                      [4096,1024], K=4096
  k_gemm_bf16<bf16><<<dim3(HH / 128, Nn / 128), 256, 0, stream>>>(
      Bn, T1T, H1, Nn, HH, Nn, b1, 1, 0, bn1_g, bn1_b, bn1_m, bn1_v);
  // T2T = (H1 @ w2)^T                                 [512,4096]
  k_gemm_bf16<bf16><<<dim3(COUT / 128, Nn / 128), 256, 0, stream>>>(
      H1, W2T, T2T, Nn, COUT, HH, nullptr, 0, 1, nullptr, nullptr, nullptr, nullptr);
  // H2 = bn2(relu(Bn @ T2 + b2))   (f32)              [4096,512], K=4096
  k_gemm_bf16<float><<<dim3(COUT / 128, Nn / 128), 256, 0, stream>>>(
      Bn, T2T, H2, Nn, COUT, Nn, b2, 1, 0, bn2_g, bn2_b, bn2_m, bn2_v);

  // ---- pool + sigmoid gate ----
  k_pool<<<(COUT + 255) / 256, 256, 0, stream>>>(H2, pooled);
  k_final<<<1, 256, 0, stream>>>(pooled, w_attn, b_attn, out);
}
